// GeneratorWithMap_8409545966256
// MI455X (gfx1250) — matrix operand order repaired and checked by reading, not compiled, not run on hardware
//
#include <hip/hip_runtime.h>
#include <math.h>

// ---------------- CDNA5 WMMA types ----------------
typedef __attribute__((ext_vector_type(16))) __bf16 v16bf;
typedef __attribute__((ext_vector_type(8)))  float  v8f;

#define DEV __device__ __forceinline__

DEV unsigned short f2bf(float f) {                 // f32 -> bf16 (RNE)
  unsigned int u = __float_as_uint(f);
  u = (u + 0x7FFFu + ((u >> 16) & 1u)) >> 16;
  return (unsigned short)u;
}
DEV float bf2f(unsigned short h) { return __uint_as_float(((unsigned int)h) << 16); }
DEV float lrelu2(float v) { return (v >= 0.f ? v : 0.2f * v) * 1.41421356237f; }

union Frag { v16bf v; unsigned int u[8]; uint4 q[2]; };

struct MlpArgs { const float* w[8]; const float* b[8]; };

// =======================================================================
// Pre-scale + convert the 8 MLP weight matrices to bf16 once, so the MLP
// WMMA K-loop is pure loads + wmma (no per-element cvt in the hot loop).
// =======================================================================
__global__ void k_prep_mlp(MlpArgs args, unsigned short* __restrict__ wbf) {
  int tid = blockIdx.x * blockDim.x + threadIdx.x;
  if (tid >= 8 * 512 * 512) return;
  int L = tid >> 18;
  int r = tid & 0x3FFFF;
  const float lrs = 0.01f * 0.044194173824159216f;  // 0.01 / sqrt(512)
  wbf[tid] = f2bf(args.w[L][r] * lrs);
}

// =======================================================================
// Style MLP: PixelNorm + 8x EqualLinear(512x512, lr_mul=0.01) + fused lrelu
// One workgroup, 16 waves. WMMA bf16, activations ping-pong through LDS.
// =======================================================================
__global__ void __launch_bounds__(512) k_mlp(const float* __restrict__ z, MlpArgs args,
                                             const unsigned short* __restrict__ wbf,
                                             float* __restrict__ wlat) {
  __shared__ __align__(16) unsigned short xbuf[16][512];
  __shared__ float part[512];
  __shared__ float red[2];
  const int tid = threadIdx.x;

  // PixelNorm over the two real batch rows
  for (int b = 0; b < 2; ++b) {
    float v = z[b * 512 + tid];
    part[tid] = v * v;
    __syncthreads();
    for (int off = 256; off > 0; off >>= 1) {
      if (tid < off) part[tid] += part[tid + off];
      __syncthreads();
    }
    if (tid == 0) red[b] = rsqrtf(part[0] / 512.f + 1e-8f);
    __syncthreads();
  }
  for (int m = 0; m < 16; ++m)
    xbuf[m][tid] = (m < 2) ? f2bf(z[m * 512 + tid] * red[m]) : (unsigned short)0;
  __syncthreads();

  const int wave = tid >> 5, lane = tid & 31;
  const bool hi = lane >= 16;
  const int l15 = lane & 15;

  for (int L = 0; L < 8; ++L) {
    const float* Bv = args.b[L];
    v8f accs[2];
    for (int nt = 0; nt < 2; ++nt) {
      const int n0 = (wave * 2 + nt) * 16;
      v8f acc = {0.f, 0.f, 0.f, 0.f, 0.f, 0.f, 0.f, 0.f};
      const unsigned short* wrow = wbf + ((size_t)L << 18) + (size_t)(n0 + l15) * 512;
      for (int kc = 0; kc < 512; kc += 32) {
        const int k0a = kc + (hi ? 8 : 0);
        const int k0b = kc + (hi ? 8 : 0);
        Frag a, bf;
        a.q[0] = *(const uint4*)(&xbuf[l15][k0a]);
        a.q[1] = *(const uint4*)(&xbuf[l15][k0a + 16]);
        bf.q[0] = *(const uint4*)(wrow + k0b);
        bf.q[1] = *(const uint4*)(wrow + k0b + 16);
        acc = __builtin_amdgcn_wmma_f32_16x16x32_bf16(false, a.v, false, bf.v,
                                                      (short)0, acc, false, false);
      }
      accs[nt] = acc;
    }
    __syncthreads();
    for (int nt = 0; nt < 2; ++nt) {
      const int n0 = (wave * 2 + nt) * 16;
      const int oc = n0 + l15;
      const float bb = Bv[oc] * 0.01f;
      v8f acc = accs[nt];
#pragma unroll
      for (int r = 0; r < 8; ++r) {
        const int m = (hi ? 8 : 0) + r;
        xbuf[m][oc] = f2bf(lrelu2(acc[r] + bb));
      }
    }
    __syncthreads();
  }
  for (int b = 0; b < 2; ++b) wlat[b * 512 + tid] = bf2f(xbuf[b][tid]);
}

// =======================================================================
// Per-layer style prep kernels
// =======================================================================
__global__ void k_style_s(const float* __restrict__ wlat, const float* __restrict__ mw,
                          const float* __restrict__ mbias, float* __restrict__ s, int IC) {
  int tid = blockIdx.x * blockDim.x + threadIdx.x;
  if (tid >= 2 * IC) return;
  int i = tid % IC, b = tid / IC;
  const float* wl = wlat + b * 512;
  const float* mr = mw + (size_t)i * 512;
  float a = 0.f;
  for (int k = 0; k < 512; ++k) a += wl[k] * mr[k];
  s[tid] = a * 0.044194173824159216f + mbias[i];
}

__global__ void k_style_d(const float* __restrict__ w, const float* __restrict__ s,
                          float* __restrict__ d, int IC, int OC) {
  int tid = blockIdx.x * blockDim.x + threadIdx.x;
  if (tid >= 2 * OC) return;
  int o = tid % OC, b = tid / OC;
  const float* wr = w + (size_t)o * IC * 9;
  const float* sb = s + b * IC;
  float acc = 0.f;
  for (int i = 0; i < IC; ++i) {
    float ss = sb[i] * sb[i];
    float aw = 0.f;
    for (int t = 0; t < 9; ++t) { float x = wr[i * 9 + t]; aw += x * x; }
    acc += aw * ss;
  }
  d[tid] = rsqrtf(acc / (float)(IC * 9) + 1e-8f);
}

// wmod[b][tap][o][i] = scale * w[o][i][tap'] * s[b][i] * d[b][o]  (bf16)
__global__ void k_prep_w(const float* __restrict__ w, const float* __restrict__ s,
                         const float* __restrict__ d, unsigned short* __restrict__ wmod,
                         int IC, int OC, int flip) {
  size_t tid = (size_t)blockIdx.x * blockDim.x + threadIdx.x;
  size_t total = (size_t)2 * 9 * OC * IC;
  if (tid >= total) return;
  int i = (int)(tid % IC);
  int o = (int)((tid / IC) % OC);
  int t = (int)((tid / ((size_t)IC * OC)) % 9);
  int b = (int)(tid / ((size_t)IC * OC * 9));
  int ts = flip ? 8 - t : t;
  float scale = rsqrtf((float)(IC * 9));
  float v = scale * w[((size_t)o * IC + i) * 9 + ts] * s[b * IC + i] * d[b * OC + o];
  wmod[tid] = f2bf(v);
}

__global__ void k_prep_rgb(const float* __restrict__ w, const float* __restrict__ s,
                           float* __restrict__ wrgb, int IC) {
  int tid = blockIdx.x * blockDim.x + threadIdx.x;
  if (tid >= 2 * IC) return;
  int i = tid % IC, b = tid / IC;
  float scale = rsqrtf((float)IC);
  for (int o = 0; o < 3; ++o)
    wrgb[((size_t)b * IC + i) * 3 + o] = scale * w[o * IC + i] * s[tid];
}

__global__ void k_const_init(const float* __restrict__ cst, unsigned short* __restrict__ act) {
  int tid = blockIdx.x * blockDim.x + threadIdx.x;
  if (tid >= 2 * 16 * 512) return;
  int c = tid % 512;
  int p = (tid / 512) % 16;
  act[tid] = f2bf(cst[c * 16 + p]);  // NCHW const -> NHWC bf16, broadcast over batch
}

// =======================================================================
// Modulated 3x3 conv as implicit GEMM on WMMA bf16.
// 4 waves / workgroup; each wave owns one 16-pixel M tile and TWO 16-wide
// N tiles (two accumulators share each A fragment -> half the global A
// traffic, 2x arithmetic intensity). The 32xIC weight panel is staged
// per-tap into LDS with coalesced b128 loads.
//   DIL=true : transposed-conv path (dilated input, flipped taps baked into
//              wmod, pad=2), output (2H+1)^2, epilogue deferred to blur.
//   FUSE=true: stylemap scale/shift + noise + FusedLeakyReLU fused here.
// Activations NHWC bf16; wmod [b][tap][o][i] bf16.
// =======================================================================
template<int IC, int OC, int HIN, bool DIL, bool FUSE>
__global__ void __launch_bounds__(128) k_conv(
    const unsigned short* __restrict__ actIn,
    const unsigned short* __restrict__ wmod,
    const float* __restrict__ smS, const float* __restrict__ smB, int smStride,
    const float* __restrict__ noise, const float* __restrict__ nw,
    const float* __restrict__ ab,
    unsigned short* __restrict__ actOut) {
  constexpr int H = HIN, W = HIN;
  constexpr int OW = DIL ? 2 * HIN + 1 : HIN;
  constexpr int NPIX = OW * OW;
  __shared__ __align__(16) unsigned short wlds[32 * IC];

  const int tid = threadIdx.x;
  const int wave = tid >> 5, lane = tid & 31;
  const bool hi = lane >= 16;
  const int l15 = lane & 15;
  const int p0 = (blockIdx.x * 4 + wave) * 16;
  const int n0 = blockIdx.y * 32;
  const int b = blockIdx.z;

  const int pA = p0 + l15;  // this lane's A-matrix row (output pixel)
  const int oy = pA / OW, ox = pA % OW;
  const bool pixOk = pA < NPIX;

  const size_t inB = (size_t)b * H * W * IC;
  const size_t wB = (size_t)b * 9 * OC * IC + (size_t)n0 * IC;

  v8f acc0 = {0.f, 0.f, 0.f, 0.f, 0.f, 0.f, 0.f, 0.f};
  v8f acc1 = {0.f, 0.f, 0.f, 0.f, 0.f, 0.f, 0.f, 0.f};

  for (int t = 0; t < 9; ++t) {
    // ---- stage this tap's 32xIC bf16 weight panel into LDS ----
    __syncthreads();
    {
      const uint4* wpan = (const uint4*)(wmod + wB + (size_t)t * OC * IC);
      uint4* dst = (uint4*)wlds;
      for (int u2 = tid; u2 < (32 * IC) / 8; u2 += 128) dst[u2] = wpan[u2];
    }
    __syncthreads();

    const int ty = t / 3, tx = t % 3;
    int iy, ix;
    bool val;
    if (DIL) {
      int uy = oy + ty - 2, ux = ox + tx - 2;
      val = pixOk && uy >= 0 && ux >= 0 && !(uy & 1) && !(ux & 1);
      iy = uy >> 1; ix = ux >> 1;
      val = val && iy < H && ix < W;
    } else {
      iy = oy + ty - 1; ix = ox + tx - 1;
      val = pixOk && iy >= 0 && iy < H && ix >= 0 && ix < W;
    }
    const unsigned short* ap = actIn + inB + ((size_t)(iy * W + ix)) * IC;
    const unsigned short* wl0 = &wlds[l15 * IC];
    const unsigned short* wl1 = &wlds[(l15 + 16) * IC];

    for (int kc = 0; kc < IC; kc += 32) {
      const int k0a = kc + (hi ? 8 : 0);
      const int k0b = kc + (hi ? 8 : 0);
      Frag a, b0, b1;
      if (val) {
        a.q[0] = *(const uint4*)(ap + k0a);
        a.q[1] = *(const uint4*)(ap + k0a + 16);
      } else {
#pragma unroll
        for (int j = 0; j < 8; ++j) a.u[j] = 0u;
      }
      b0.q[0] = *(const uint4*)(wl0 + k0b);
      b0.q[1] = *(const uint4*)(wl0 + k0b + 16);
      b1.q[0] = *(const uint4*)(wl1 + k0b);
      b1.q[1] = *(const uint4*)(wl1 + k0b + 16);
      acc0 = __builtin_amdgcn_wmma_f32_16x16x32_bf16(false, a.v, false, b0.v,
                                                     (short)0, acc0, false, false);
      acc1 = __builtin_amdgcn_wmma_f32_16x16x32_bf16(false, a.v, false, b1.v,
                                                     (short)0, acc1, false, false);
    }
  }

  // D layout: lanes 0-15 -> rows 0..7 (vgpr r), col n0+lane; lanes 16-31 -> rows 8..15
#pragma unroll
  for (int half = 0; half < 2; ++half) {
    const int oc = n0 + l15 + 16 * half;
    const v8f acc = half ? acc1 : acc0;
#pragma unroll
    for (int r = 0; r < 8; ++r) {
      const int pm = p0 + (hi ? 8 + r : r);
      if (pm < NPIX) {
        float v = acc[r];
        if constexpr (FUSE) {
          v = v * smS[b * smStride + pm] + smB[b * smStride + pm];
          v += nw[0] * noise[pm];
          v = lrelu2(v + ab[oc]);
        }
        actOut[((size_t)b * NPIX + pm) * OC + oc] = f2bf(v);
      }
    }
  }
}

// 4x4 blur (pad 1,1) on the (2H+1)^2 transposed-conv output + fused epilogue
// 2 channels per thread: one dword load per tap, packed bf16x2 store.
template<int OC, int HOUT>
__global__ void k_blur_epi(const unsigned short* __restrict__ pre,
                           const float* __restrict__ smS, const float* __restrict__ smB,
                           int smStride,
                           const float* __restrict__ noise, const float* __restrict__ nw,
                           const float* __restrict__ ab,
                           unsigned short* __restrict__ actOut) {
  constexpr int P = HOUT + 1;
  constexpr int OC2 = OC / 2;
  const size_t tid = (size_t)blockIdx.x * blockDim.x + threadIdx.x;
  const size_t total = (size_t)2 * HOUT * HOUT * OC2;
  if (tid >= total) return;
  const int c = 2 * (int)(tid % OC2);
  const int pm = (int)((tid / OC2) % ((size_t)HOUT * HOUT));
  const int b = (int)(tid / ((size_t)OC2 * HOUT * HOUT));
  const int y = pm / HOUT, x = pm % HOUT;
  const float k1[4] = {1.f, 3.f, 3.f, 1.f};
  float a0 = 0.f, a1 = 0.f;
#pragma unroll
  for (int ky = 0; ky < 4; ++ky) {
    const int iy = y - 1 + ky;
    if (iy < 0 || iy >= P) continue;
#pragma unroll
    for (int kx = 0; kx < 4; ++kx) {
      const int ix = x - 1 + kx;
      if (ix < 0 || ix >= P) continue;
      const float cf = k1[ky] * k1[kx] * (1.f / 16.f);
      unsigned int pk = *(const unsigned int*)(pre + (((size_t)b * P + iy) * P + ix) * OC + c);
      a0 += cf * bf2f((unsigned short)pk);
      a1 += cf * bf2f((unsigned short)(pk >> 16));
    }
  }
  const float ss = smS[b * smStride + pm], sb = smB[b * smStride + pm];
  const float nz = nw[0] * noise[pm];
  float v0 = lrelu2(a0 * ss + sb + nz + ab[c]);
  float v1 = lrelu2(a1 * ss + sb + nz + ab[c + 1]);
  unsigned int packed = (unsigned int)f2bf(v0) | ((unsigned int)f2bf(v1) << 16);
  *(unsigned int*)(actOut + ((size_t)b * HOUT * HOUT + pm) * OC + c) = packed;
}

// =======================================================================
// toRGB: modulated 1x1 conv (no demod) + bias (+ upsampled skip)
// =======================================================================
template<int IC, int R, bool SK>
__global__ void k_torgb(const unsigned short* __restrict__ act,
                        const float* __restrict__ wrgb,   // [B][IC][3]
                        const float* __restrict__ bias,   // [3]
                        const float* __restrict__ skipIn, // [B][3][R/2][R/2]
                        float* __restrict__ outRgb) {     // [B][3][R][R]
  const int tid = blockIdx.x * blockDim.x + threadIdx.x;
  if (tid >= 2 * R * R) return;
  const int pm = tid % (R * R);
  const int b = tid / (R * R);
  float a0 = 0.f, a1 = 0.f, a2 = 0.f;
  const unsigned short* ap = act + ((size_t)b * R * R + pm) * IC;
  const float* wp = wrgb + (size_t)b * IC * 3;
  for (int i = 0; i < IC; i += 2) {
    unsigned int pk = *(const unsigned int*)(ap + i);
    float x0 = bf2f((unsigned short)pk), x1 = bf2f((unsigned short)(pk >> 16));
    a0 += x0 * wp[i * 3 + 0] + x1 * wp[(i + 1) * 3 + 0];
    a1 += x0 * wp[i * 3 + 1] + x1 * wp[(i + 1) * 3 + 1];
    a2 += x0 * wp[i * 3 + 2] + x1 * wp[(i + 1) * 3 + 2];
  }
  float o[3] = {a0 + bias[0], a1 + bias[1], a2 + bias[2]};
  if constexpr (SK) {
    constexpr int Rh = R / 2;
    const int y = pm / R, x = pm % R;
    const float k1[4] = {1.f, 3.f, 3.f, 1.f};
#pragma unroll
    for (int o3 = 0; o3 < 3; ++o3) {
      float s = 0.f;
      for (int ky = 0; ky < 4; ++ky) {
        int t = y + ky - 2;
        if (t < 0 || t > 2 * Rh - 2 || (t & 1)) continue;
        int iy = t >> 1;
        for (int kx = 0; kx < 4; ++kx) {
          int u = x + kx - 2;
          if (u < 0 || u > 2 * Rh - 2 || (u & 1)) continue;
          int ix = u >> 1;
          s += k1[ky] * k1[kx] * (1.f / 16.f) *
               skipIn[(((size_t)b * 3 + o3) * Rh + iy) * Rh + ix];
        }
      }
      o[o3] += s;
    }
  }
#pragma unroll
  for (int o3 = 0; o3 < 3; ++o3)
    outRgb[(((size_t)b * 3 + o3) * R * R) + pm] = o[o3];
}

// =======================================================================
// Mesh ResBlock (3->3->OC, 1x1 skip) with inline bilinear resize of mesh
// =======================================================================
template<int OC, int RES>
__global__ void k_resblock(const float* __restrict__ mesh,  // [2][3][256][256]
                           const float* __restrict__ c1w, const float* __restrict__ c1b,
                           const float* __restrict__ c2w, const float* __restrict__ c2b,
                           const float* __restrict__ skw,
                           float* __restrict__ sm) {        // [2][OC][RES][RES]
  const int tid = blockIdx.x * blockDim.x + threadIdx.x;
  if (tid >= 2 * RES * RES) return;
  const int pm = tid % (RES * RES);
  const int b = tid / (RES * RES);
  const int y = pm / RES, x = pm % RES;
  const float scale = 256.f / RES;
  const float* mb = mesh + (size_t)b * 3 * 65536;

  float win[3][5][5];
  for (int dy = -2; dy <= 2; ++dy)
    for (int dx = -2; dx <= 2; ++dx) {
      int fy = y + dy, fx = x + dx;
      if (fy < 0 || fy >= RES || fx < 0 || fx >= RES) {
        win[0][dy + 2][dx + 2] = win[1][dy + 2][dx + 2] = win[2][dy + 2][dx + 2] = 0.f;
      } else {
        float sy = fminf(fmaxf((fy + 0.5f) * scale - 0.5f, 0.f), 255.f);
        float sx = fminf(fmaxf((fx + 0.5f) * scale - 0.5f, 0.f), 255.f);
        int y0 = (int)sy, x0 = (int)sx;
        int y1 = min(y0 + 1, 255), x1 = min(x0 + 1, 255);
        float wy = sy - y0, wx = sx - x0;
        for (int c = 0; c < 3; ++c) {
          const float* mp = mb + (size_t)c * 65536;
          float v00 = mp[y0 * 256 + x0], v01 = mp[y0 * 256 + x1];
          float v10 = mp[y1 * 256 + x0], v11 = mp[y1 * 256 + x1];
          win[c][dy + 2][dx + 2] = v00 * (1 - wy) * (1 - wx) + v01 * (1 - wy) * wx +
                                   v10 * wy * (1 - wx) + v11 * wy * wx;
        }
      }
    }
  const float s1 = rsqrtf(27.f), s2 = rsqrtf(27.f), s3 = rsqrtf(3.f);
  float h1[3][3][3];
  for (int py = 0; py < 3; ++py)
    for (int px = 0; px < 3; ++px)
      for (int co = 0; co < 3; ++co) {
        float a = 0.f;
        for (int ci = 0; ci < 3; ++ci)
          for (int ky = 0; ky < 3; ++ky)
            for (int kx = 0; kx < 3; ++kx)
              a += win[ci][py + ky][px + kx] * c1w[((co * 3 + ci) * 3 + ky) * 3 + kx];
        int hy = y + py - 1, hx = x + px - 1;
        h1[co][py][px] = (hy < 0 || hy >= RES || hx < 0 || hx >= RES)
                             ? 0.f
                             : lrelu2(a * s1 + c1b[co]);
      }
  for (int o = 0; o < OC; ++o) {
    float a = 0.f;
    for (int ci = 0; ci < 3; ++ci)
      for (int ky = 0; ky < 3; ++ky)
        for (int kx = 0; kx < 3; ++kx)
          a += h1[ci][ky][kx] * c2w[((o * 3 + ci) * 3 + ky) * 3 + kx];
    float h = lrelu2(a * s2 + c2b[o]);
    float sk = 0.f;
    for (int ci = 0; ci < 3; ++ci) sk += win[ci][2][2] * skw[o * 3 + ci];
    sm[(((size_t)b * OC + o) * RES * RES) + pm] = (h + sk * s3) * 0.70710678118f;
  }
}

// =======================================================================
// Host-side orchestration
// =======================================================================
static inline const float* FIN(void* const* d_in, int i) { return (const float*)d_in[i]; }

struct Bufs {
  float *wlat, *sbuf, *dbuf, *wrgb, *sm;
  unsigned short *wmlp, *wmod, *actA, *actB, *pre;
  float *rgbA, *rgbB;
};

template<int IC, int OC, int HIN, bool DIL>
static void conv_layer(void* const* d_in, int l, const unsigned short* aIn,
                       unsigned short* aOut, Bufs& B, int smC0, int smTot, int res_sm,
                       const float* noise, hipStream_t st) {
  const float* w  = FIN(d_in, 18 + 5 * l + 0);
  const float* mw = FIN(d_in, 18 + 5 * l + 1);
  const float* mb = FIN(d_in, 18 + 5 * l + 2);
  const float* nw = FIN(d_in, 18 + 5 * l + 3);
  const float* ab = FIN(d_in, 18 + 5 * l + 4);
  k_style_s<<<dim3((2 * IC + 127) / 128), 128, 0, st>>>(B.wlat, mw, mb, B.sbuf, IC);
  k_style_d<<<dim3((2 * OC + 127) / 128), 128, 0, st>>>(w, B.sbuf, B.dbuf, IC, OC);
  {
    size_t tot = (size_t)2 * 9 * OC * IC;
    k_prep_w<<<dim3((unsigned)((tot + 255) / 256)), 256, 0, st>>>(w, B.sbuf, B.dbuf,
                                                                  B.wmod, IC, OC, DIL ? 1 : 0);
  }
  const int smStride = smTot * res_sm * res_sm;
  const float* smS = B.sm + (size_t)smC0 * res_sm * res_sm;
  const float* smB2 = smS + (size_t)res_sm * res_sm;
  if constexpr (DIL) {
    constexpr int OWp = 2 * HIN + 1;
    const int mt = (OWp * OWp + 63) / 64;
    k_conv<IC, OC, HIN, true, false><<<dim3(mt, OC / 32, 2), 128, 0, st>>>(
        aIn, B.wmod, nullptr, nullptr, 0, nullptr, nullptr, nullptr, B.pre);
    constexpr int HO = 2 * HIN;
    size_t tot = (size_t)2 * HO * HO * (OC / 2);
    k_blur_epi<OC, HO><<<dim3((unsigned)((tot + 255) / 256)), 256, 0, st>>>(
        B.pre, smS, smB2, smStride, noise, nw, ab, aOut);
  } else {
    const int mt = (HIN * HIN + 63) / 64;
    k_conv<IC, OC, HIN, false, true><<<dim3(mt, OC / 32, 2), 128, 0, st>>>(
        aIn, B.wmod, smS, smB2, smStride, noise, nw, ab, aOut);
  }
}

template<int IC, int R, bool SK>
static void rgb_layer(void* const* d_in, int j, const unsigned short* act,
                      const float* skipIn, float* outRgb, Bufs& B, hipStream_t st) {
  const float* w = FIN(d_in, 83 + 4 * j + 0);
  const float* mw = FIN(d_in, 83 + 4 * j + 1);
  const float* mb = FIN(d_in, 83 + 4 * j + 2);
  const float* bias = FIN(d_in, 83 + 4 * j + 3);
  k_style_s<<<dim3((2 * IC + 127) / 128), 128, 0, st>>>(B.wlat, mw, mb, B.sbuf, IC);
  k_prep_rgb<<<dim3((2 * IC + 127) / 128), 128, 0, st>>>(w, B.sbuf, B.wrgb, IC);
  int tot = 2 * R * R;
  k_torgb<IC, R, SK><<<dim3((tot + 127) / 128), 128, 0, st>>>(act, B.wrgb, bias, skipIn,
                                                              outRgb);
}

template<int OC, int RES>
static void resblock_layer(void* const* d_in, int k, const float* mesh, Bufs& B,
                           hipStream_t st) {
  int tot = 2 * RES * RES;
  k_resblock<OC, RES><<<dim3((tot + 63) / 64), 64, 0, st>>>(
      mesh, FIN(d_in, 111 + 5 * k + 0), FIN(d_in, 111 + 5 * k + 1),
      FIN(d_in, 111 + 5 * k + 2), FIN(d_in, 111 + 5 * k + 3), FIN(d_in, 111 + 5 * k + 4),
      B.sm);
}

extern "C" void kernel_launch(void* const* d_in, const int* in_sizes, int n_in,
                              void* d_out, int out_size, void* d_ws, size_t ws_size,
                              hipStream_t stream) {
  (void)in_sizes; (void)n_in; (void)out_size; (void)ws_size;
  char* ws = (char*)d_ws;
  size_t off = 0;
  auto carve = [&](size_t n) -> char* {
    char* p = ws + off;
    off += (n + 255) & ~(size_t)255;
    return p;
  };
  Bufs B;
  B.wlat = (float*)carve(2 * 512 * 4);
  B.sbuf = (float*)carve(2 * 512 * 4);
  B.dbuf = (float*)carve(2 * 512 * 4);
  B.wrgb = (float*)carve(2 * 512 * 3 * 4);
  B.sm   = (float*)carve((size_t)2 * 4 * 256 * 256 * 4);
  B.wmlp = (unsigned short*)carve((size_t)8 * 512 * 512 * 2);
  B.wmod = (unsigned short*)carve((size_t)2 * 9 * 512 * 512 * 2);
  B.actA = (unsigned short*)carve((size_t)2 * 256 * 256 * 128 * 2);
  B.actB = (unsigned short*)carve((size_t)2 * 256 * 256 * 128 * 2);
  B.pre  = (unsigned short*)carve((size_t)2 * 257 * 257 * 128 * 2);
  B.rgbA = (float*)carve((size_t)2 * 3 * 256 * 256 * 4);
  B.rgbB = (float*)carve((size_t)2 * 3 * 256 * 256 * 4);

  const float* z = FIN(d_in, 0);
  const float* mesh = FIN(d_in, 1);
  MlpArgs ma;
  for (int i = 0; i < 8; ++i) { ma.w[i] = FIN(d_in, 2 + 2 * i); ma.b[i] = FIN(d_in, 3 + 2 * i); }
  k_prep_mlp<<<dim3((8 * 512 * 512 + 255) / 256), 256, 0, stream>>>(ma, B.wmlp);
  k_mlp<<<1, 512, 0, stream>>>(z, ma, B.wmlp, B.wlat);
  k_const_init<<<dim3((2 * 16 * 512 + 255) / 256), 256, 0, stream>>>(FIN(d_in, 146), B.actA);

  // conv0 @ res 4  (const -> actB)
  resblock_layer<2, 4>(d_in, 0, mesh, B, stream);
  conv_layer<512, 512, 4, false>(d_in, 0, B.actA, B.actB, B, 0, 2, 4, FIN(d_in, 147), stream);
  rgb_layer<512, 4, false>(d_in, 0, B.actB, nullptr, B.rgbA, B, stream);

  // block 0: res 8
  resblock_layer<4, 8>(d_in, 1, mesh, B, stream);
  conv_layer<512, 512, 4, true >(d_in, 1, B.actB, B.actA, B, 0, 4, 8, FIN(d_in, 148), stream);
  conv_layer<512, 512, 8, false>(d_in, 2, B.actA, B.actB, B, 2, 4, 8, FIN(d_in, 149), stream);
  rgb_layer<512, 8, true>(d_in, 1, B.actB, B.rgbA, B.rgbB, B, stream);

  // block 1: res 16
  resblock_layer<4, 16>(d_in, 2, mesh, B, stream);
  conv_layer<512, 512, 8,  true >(d_in, 3, B.actB, B.actA, B, 0, 4, 16, FIN(d_in, 150), stream);
  conv_layer<512, 512, 16, false>(d_in, 4, B.actA, B.actB, B, 2, 4, 16, FIN(d_in, 151), stream);
  rgb_layer<512, 16, true>(d_in, 2, B.actB, B.rgbB, B.rgbA, B, stream);

  // block 2: res 32
  resblock_layer<4, 32>(d_in, 3, mesh, B, stream);
  conv_layer<512, 512, 16, true >(d_in, 5, B.actB, B.actA, B, 0, 4, 32, FIN(d_in, 152), stream);
  conv_layer<512, 512, 32, false>(d_in, 6, B.actA, B.actB, B, 2, 4, 32, FIN(d_in, 153), stream);
  rgb_layer<512, 32, true>(d_in, 3, B.actB, B.rgbA, B.rgbB, B, stream);

  // block 3: res 64
  resblock_layer<4, 64>(d_in, 4, mesh, B, stream);
  conv_layer<512, 512, 32, true >(d_in, 7, B.actB, B.actA, B, 0, 4, 64, FIN(d_in, 154), stream);
  conv_layer<512, 512, 64, false>(d_in, 8, B.actA, B.actB, B, 2, 4, 64, FIN(d_in, 155), stream);
  rgb_layer<512, 64, true>(d_in, 4, B.actB, B.rgbB, B.rgbA, B, stream);

  // block 4: res 128 (512 -> 256)
  resblock_layer<4, 128>(d_in, 5, mesh, B, stream);
  conv_layer<512, 256, 64,  true >(d_in, 9,  B.actB, B.actA, B, 0, 4, 128, FIN(d_in, 156), stream);
  conv_layer<256, 256, 128, false>(d_in, 10, B.actA, B.actB, B, 2, 4, 128, FIN(d_in, 157), stream);
  rgb_layer<256, 128, true>(d_in, 5, B.actB, B.rgbA, B.rgbB, B, stream);

  // block 5: res 256 (256 -> 128), final toRGB writes d_out [2,3,256,256] f32
  resblock_layer<4, 256>(d_in, 6, mesh, B, stream);
  conv_layer<256, 128, 128, true >(d_in, 11, B.actB, B.actA, B, 0, 4, 256, FIN(d_in, 158), stream);
  conv_layer<128, 128, 256, false>(d_in, 12, B.actA, B.actB, B, 2, 4, 256, FIN(d_in, 159), stream);
  rgb_layer<128, 256, true>(d_in, 6, B.actB, B.rgbB, (float*)d_out, B, stream);
}
